// UnifiedTransformerALU_62380105007484
// MI455X (gfx1250) — matrix operand order repaired and checked by reading, compiled, not run on hardware
//
#include <hip/hip_runtime.h>
#include <hip/hip_bf16.h>
#include <math.h>

#define NB 33
#define NOPS 19
#define D 128
#define H 512
#define B 8
#define S 1024
#define BS (B * S) /* 8192 rows */

typedef __bf16 bf16_t;
typedef bf16_t v16bf __attribute__((ext_vector_type(16)));
typedef float v8f __attribute__((ext_vector_type(8)));
typedef unsigned int u32x4 __attribute__((ext_vector_type(4)));
typedef int i32x8 __attribute__((ext_vector_type(8)));
typedef int i32x4 __attribute__((ext_vector_type(4)));

union frag_u {
  v16bf v;
  uint4 q[2];
};

// ---------------------------------------------------------------------------
// WMMA: D = A(16x32 bf16) * B(32x16 bf16) + C(16x16 f32)
// ---------------------------------------------------------------------------
__device__ __forceinline__ v8f wmma_bf16(v16bf a, v16bf b, v8f c) {
  return __builtin_amdgcn_wmma_f32_16x16x32_bf16(false, a, false, b, (short)0,
                                                 c, false, false);
}

// A fragment (16x32) from row-major bf16 [.. x ld]:
// lane group g: elements 0..7 <-> k0+g*8+0..7 ; 8..15 <-> k0+16+g*8+0..7
__device__ __forceinline__ v16bf load_a_bf16(const bf16_t* A, int ld, int row0,
                                             int k0, int lane) {
  int g = lane >> 4, m = lane & 15;
  const bf16_t* p = A + (size_t)(row0 + m) * ld + k0 + g * 8;
  frag_u u;
  u.q[0] = *(const uint4*)p;
  u.q[1] = *(const uint4*)(p + 16);
  return u.v;
}

// B fragment (32x16) where logical B[k][n] = Src[n][k] (Src row-major, ld=K):
// lane group g: elements 0..7 <-> k0+g*8+0..7 ; 8..15 <-> k0+16+g*8+0..7
__device__ __forceinline__ v16bf load_bT_bf16(const bf16_t* Src, int ld, int k0,
                                              int n0, int lane) {
  int g = lane >> 4, n = lane & 15;
  const bf16_t* p = Src + (size_t)(n0 + n) * ld + k0 + g * 8;
  frag_u u;
  u.q[0] = *(const uint4*)p;
  u.q[1] = *(const uint4*)(p + 16);
  return u.v;
}

// ---------------------------------------------------------------------------
// TDM: DMA a 2D bf16 tile (tile_d1 rows x tile_d0 elems) into LDS.
// D# layout per CDNA5 ISA ch.8: group0 {count,lds_addr,global_addr,type=2},
// group1 {data_size=2B, tensor dims, tile dims, dim0 stride}. 2D -> g2/g3/g4
// zero-filled. This toolchain's builtin takes 6 args (extra i32x8 group).
// ---------------------------------------------------------------------------
__device__ __forceinline__ void tdm_load_tile_2d(void* lds_dst,
                                                 const void* gsrc,
                                                 unsigned tensor_d0,
                                                 unsigned tensor_d1,
                                                 unsigned tile_d0,
                                                 unsigned tile_d1,
                                                 unsigned d0_stride) {
  unsigned long long ga = (unsigned long long)(uintptr_t)gsrc;
  unsigned lds = (unsigned)(uintptr_t)lds_dst;
  u32x4 g0;
  g0[0] = 1u;  // count=1, user mode, gather off
  g0[1] = lds; // lds_addr (bytes)
  g0[2] = (unsigned)(ga & 0xFFFFFFFFu);               // global_addr[31:0]
  g0[3] = (unsigned)((ga >> 32) & 0x01FFFFFFu) |      // global_addr[56:32]
          (2u << 30);                                 // type=2 ("image")
  i32x8 g1;
  g1[0] = (int)(1u << 16);  // workgroup_mask=0, data_size=1 (2 bytes)
  g1[1] = (int)((tensor_d0 & 0xFFFFu) << 16);         // atomic_barrier=0
  g1[2] = (int)((tensor_d0 >> 16) | ((tensor_d1 & 0xFFFFu) << 16));
  g1[3] = (int)((tensor_d1 >> 16) | (tile_d0 << 16));
  g1[4] = (int)(tile_d1 & 0xFFFFu);  // tile_dim1; tile_dim2=0
  g1[5] = (int)d0_stride;            // tensor_dim0_stride[31:0]
  g1[6] = 0;                         // stride hi + dim1_stride lo
  g1[7] = 0;                         // dim1_stride hi
  i32x4 z4 = {0, 0, 0, 0};
  i32x8 z8 = {0, 0, 0, 0, 0, 0, 0, 0};
  __builtin_amdgcn_tensor_load_to_lds(g0, g1, z4, z4, z8, 0);
}

// ---------------------------------------------------------------------------
// 1) routing: op = argmax(h[0,0,:19]); flags = masks[blk][op]
// ---------------------------------------------------------------------------
__global__ void route_kernel(const float* __restrict__ h,
                             const unsigned char* __restrict__ am,
                             const unsigned char* __restrict__ fm, int blk,
                             int* __restrict__ ctl) {
  if (threadIdx.x == 0) {
    int op = 0;
    float best = h[0];
    for (int i = 1; i < NOPS; ++i) {
      float v = h[i];
      if (v > best) { best = v; op = i; }
    }
    ctl[0] = op;
    ctl[1] = am[blk * NOPS + op] ? 1 : 0;
    ctl[2] = fm[blk * NOPS + op] ? 1 : 0;
  }
}

// ---------------------------------------------------------------------------
// 2) h (f32) -> hb (bf16), 4 elems/thread
// ---------------------------------------------------------------------------
__global__ void convert_h_kernel(const float* __restrict__ h,
                                 bf16_t* __restrict__ hb) {
  int i = blockIdx.x * blockDim.x + threadIdx.x;  // float4 index
  float4 f = ((const float4*)h)[i];
  union { bf16_t b[4]; uint2 u; } pk;
  pk.b[0] = (bf16_t)f.x;
  pk.b[1] = (bf16_t)f.y;
  pk.b[2] = (bf16_t)f.z;
  pk.b[3] = (bf16_t)f.w;
  ((uint2*)hb)[i] = pk.u;
}

// ---------------------------------------------------------------------------
// 3) convert + transpose the 6 routed weight matrices to bf16 once per block
// ---------------------------------------------------------------------------
__global__ void convert_w_kernel(const float* __restrict__ wq,
                                 const float* __restrict__ wk,
                                 const float* __restrict__ wv,
                                 const float* __restrict__ wo,
                                 const float* __restrict__ w1,
                                 const float* __restrict__ w2, int blk,
                                 const int* __restrict__ ctl,
                                 bf16_t* __restrict__ wT) {
  if (!ctl[1] && !ctl[2]) return;
  int op = ctl[0];
  size_t dd = (size_t)((size_t)blk * NOPS + op);
  int i = blockIdx.x * blockDim.x + threadIdx.x;  // 0 .. 196607
  const int DDSZ = D * D, DHSZ = D * H;
  if (i < 4 * DDSZ) {
    int mat = i / DDSZ, r = i % DDSZ;
    int n = r / D, k = r % D;  // wT[n][k] = W[k][n]
    const float* W =
        (mat == 0 ? wq : mat == 1 ? wk : mat == 2 ? wv : wo) + dd * DDSZ;
    wT[(size_t)mat * DDSZ + r] = (bf16_t)W[k * D + n];
  } else if (i < 4 * DDSZ + DHSZ) {
    int r = i - 4 * DDSZ;      // w1T is HxD
    int n = r / D, k = r % D;  // w1T[n][k] = w1[k][n]
    wT[(size_t)4 * DDSZ + r] = (bf16_t)(w1 + dd * DHSZ)[(size_t)k * H + n];
  } else {
    int r = i - 4 * DDSZ - DHSZ;  // w2T is DxH
    int n = r / H, k = r % H;     // w2T[n][k] = w2[k][n]
    wT[(size_t)4 * DDSZ + DHSZ + r] =
        (bf16_t)(w2 + dd * DHSZ)[(size_t)k * D + n];
  }
}

// ---------------------------------------------------------------------------
// 4) QKV: wave computes 16x64 (4 tiles), depth-1 software pipeline over K
// ---------------------------------------------------------------------------
__global__ void qkv_kernel(const bf16_t* __restrict__ hb,
                           const bf16_t* __restrict__ wT,
                           const int* __restrict__ ctl,
                           bf16_t* __restrict__ qb, bf16_t* __restrict__ kb,
                           bf16_t* __restrict__ vTb) {
  if (!ctl[1]) return;
  int lane = threadIdx.x & 31;
  int wave = blockIdx.x * (blockDim.x >> 5) + (threadIdx.x >> 5);
  int ng = wave & 1;
  int mt = (wave >> 1) & 511;
  int mat = wave >> 10;  // 0=Q 1=K 2=V
  const bf16_t* w = wT + (size_t)mat * D * D;
  int row0 = mt * 16, nbase = ng * 64;
  v8f c[4] = {};
  v16bf a = load_a_bf16(hb, D, row0, 0, lane);
  v16bf b0 = load_bT_bf16(w, D, 0, nbase, lane);
  v16bf b1 = load_bT_bf16(w, D, 0, nbase + 16, lane);
  v16bf b2 = load_bT_bf16(w, D, 0, nbase + 32, lane);
  v16bf b3 = load_bT_bf16(w, D, 0, nbase + 48, lane);
#pragma unroll
  for (int kc = 0; kc < D; kc += 32) {
    v16bf an, n0, n1, n2, n3;
    if (kc + 32 < D) {  // prefetch next K-chunk while current WMMAs issue
      an = load_a_bf16(hb, D, row0, kc + 32, lane);
      n0 = load_bT_bf16(w, D, kc + 32, nbase, lane);
      n1 = load_bT_bf16(w, D, kc + 32, nbase + 16, lane);
      n2 = load_bT_bf16(w, D, kc + 32, nbase + 32, lane);
      n3 = load_bT_bf16(w, D, kc + 32, nbase + 48, lane);
    }
    c[0] = wmma_bf16(a, b0, c[0]);
    c[1] = wmma_bf16(a, b1, c[1]);
    c[2] = wmma_bf16(a, b2, c[2]);
    c[3] = wmma_bf16(a, b3, c[3]);
    a = an; b0 = n0; b1 = n1; b2 = n2; b3 = n3;
  }
  int g = lane >> 4, n = lane & 15;
  const float scale = 0.08838834764831845f;  // 1/sqrt(D) folded into Q
  if (mat == 0) {
#pragma unroll
    for (int t = 0; t < 4; ++t)
#pragma unroll
      for (int r = 0; r < 8; ++r)
        qb[(size_t)(row0 + r + g * 8) * D + nbase + t * 16 + n] =
            (bf16_t)(c[t][r] * scale);
  } else if (mat == 1) {
#pragma unroll
    for (int t = 0; t < 4; ++t)
#pragma unroll
      for (int r = 0; r < 8; ++r)
        kb[(size_t)(row0 + r + g * 8) * D + nbase + t * 16 + n] =
            (bf16_t)c[t][r];
  } else {
    int batch = row0 >> 10, key0 = row0 & (S - 1);
    bf16_t* vt = vTb + (size_t)batch * D * S;
#pragma unroll
    for (int t = 0; t < 4; ++t)
#pragma unroll
      for (int r = 0; r < 8; ++r)
        vt[(size_t)(nbase + t * 16 + n) * S + key0 + r + g * 8] =
            (bf16_t)c[t][r];
  }
}

// ---------------------------------------------------------------------------
// 5) flash attention: wave per (batch, 16-query block); 32 keys/iter
// ---------------------------------------------------------------------------
__global__ void attn_kernel(const bf16_t* __restrict__ qb,
                            const bf16_t* __restrict__ kb,
                            const bf16_t* __restrict__ vTb,
                            const int* __restrict__ ctl,
                            bf16_t* __restrict__ ob) {
  if (!ctl[1]) return;
  __shared__ float s_scores[8][16][32];
  __shared__ bf16_t s_probs[8][16][32];
  __shared__ float s_max[8][16];
  __shared__ float s_sum[8][16];
  __shared__ float s_corr[8][16];
  int lane = threadIdx.x & 31;
  int w = threadIdx.x >> 5;
  int gw = blockIdx.x * 8 + w;  // 512 waves
  int batch = gw >> 6;
  int qb16 = gw & 63;
  const bf16_t* qbase = qb + (size_t)batch * S * D;
  const bf16_t* kbase = kb + (size_t)batch * S * D;
  const bf16_t* vtb = vTb + (size_t)batch * D * S;
  int row0 = qb16 * 16;
  int g = lane >> 4, nl = lane & 15;

  v16bf qa[4];
#pragma unroll
  for (int i = 0; i < 4; ++i) qa[i] = load_a_bf16(qbase, D, row0, i * 32, lane);

  v8f acc[8] = {};
  if (lane < 16) { s_max[w][lane] = -1e30f; s_sum[w][lane] = 0.f; }

  for (int key0 = 0; key0 < S; key0 += 32) {
    v8f s0 = {}, s1 = {};
#pragma unroll
    for (int i = 0; i < 4; ++i) {
      s0 = wmma_bf16(qa[i], load_bT_bf16(kbase, D, i * 32, key0, lane), s0);
      s1 = wmma_bf16(qa[i], load_bT_bf16(kbase, D, i * 32, key0 + 16, lane), s1);
    }
#pragma unroll
    for (int r = 0; r < 8; ++r) {
      s_scores[w][r + g * 8][nl] = s0[r];
      s_scores[w][r + g * 8][16 + nl] = s1[r];
    }
    if (lane < 16) {  // per-row online softmax
      float om = s_max[w][lane];
      float mx = om;
#pragma unroll
      for (int j = 0; j < 32; ++j) mx = fmaxf(mx, s_scores[w][lane][j]);
      float corr = __expf(om - mx);
      float sum = s_sum[w][lane] * corr;
#pragma unroll
      for (int j = 0; j < 32; ++j) {
        float p = __expf(s_scores[w][lane][j] - mx);
        s_probs[w][lane][j] = (bf16_t)p;
        sum += p;
      }
      s_max[w][lane] = mx;
      s_sum[w][lane] = sum;
      s_corr[w][lane] = corr;
    }
    float cr[8];
#pragma unroll
    for (int r = 0; r < 8; ++r) cr[r] = s_corr[w][r + g * 8];
#pragma unroll
    for (int t = 0; t < 8; ++t)
#pragma unroll
      for (int r = 0; r < 8; ++r) acc[t][r] *= cr[r];
    v16bf pa = load_a_bf16(&s_probs[w][0][0], 32, 0, 0, lane);
#pragma unroll
    for (int t = 0; t < 8; ++t)
      acc[t] = wmma_bf16(pa, load_bT_bf16(vtb, S, key0, t * 16, lane), acc[t]);
  }
  float inv[8];
#pragma unroll
  for (int r = 0; r < 8; ++r) inv[r] = 1.f / s_sum[w][r + g * 8];
  bf16_t* obase = ob + (size_t)batch * S * D;
#pragma unroll
  for (int t = 0; t < 8; ++t)
#pragma unroll
    for (int r = 0; r < 8; ++r)
      obase[(size_t)(row0 + r + g * 8) * D + t * 16 + nl] =
          (bf16_t)(acc[t][r] * inv[r]);
}

// ---------------------------------------------------------------------------
// 6) out-proj: h = O @ wo (routed); dual store f32 h + bf16 hb
// ---------------------------------------------------------------------------
__global__ void outproj_kernel(const bf16_t* __restrict__ ob,
                               const bf16_t* __restrict__ wT,
                               const int* __restrict__ ctl,
                               float* __restrict__ h, bf16_t* __restrict__ hb) {
  if (!ctl[1]) return;
  const bf16_t* w = wT + (size_t)3 * D * D;  // woT
  int lane = threadIdx.x & 31;
  int wave = blockIdx.x * (blockDim.x >> 5) + (threadIdx.x >> 5);
  int ng = wave & 1, mt = wave >> 1;  // 1024 waves
  int row0 = mt * 16, nbase = ng * 64;
  v8f c[4] = {};
  v16bf a = load_a_bf16(ob, D, row0, 0, lane);
  v16bf b0 = load_bT_bf16(w, D, 0, nbase, lane);
  v16bf b1 = load_bT_bf16(w, D, 0, nbase + 16, lane);
  v16bf b2 = load_bT_bf16(w, D, 0, nbase + 32, lane);
  v16bf b3 = load_bT_bf16(w, D, 0, nbase + 48, lane);
#pragma unroll
  for (int kc = 0; kc < D; kc += 32) {
    v16bf an, n0, n1, n2, n3;
    if (kc + 32 < D) {
      an = load_a_bf16(ob, D, row0, kc + 32, lane);
      n0 = load_bT_bf16(w, D, kc + 32, nbase, lane);
      n1 = load_bT_bf16(w, D, kc + 32, nbase + 16, lane);
      n2 = load_bT_bf16(w, D, kc + 32, nbase + 32, lane);
      n3 = load_bT_bf16(w, D, kc + 32, nbase + 48, lane);
    }
    c[0] = wmma_bf16(a, b0, c[0]);
    c[1] = wmma_bf16(a, b1, c[1]);
    c[2] = wmma_bf16(a, b2, c[2]);
    c[3] = wmma_bf16(a, b3, c[3]);
    a = an; b0 = n0; b1 = n1; b2 = n2; b3 = n3;
  }
  int g = lane >> 4, n = lane & 15;
#pragma unroll
  for (int t = 0; t < 4; ++t)
#pragma unroll
    for (int r = 0; r < 8; ++r) {
      size_t idx = (size_t)(row0 + r + g * 8) * D + nbase + t * 16 + n;
      h[idx] = c[t][r];
      hb[idx] = (bf16_t)c[t][r];
    }
}

// ---------------------------------------------------------------------------
// 7) fused FFN with TDM weight staging: one wave DMAs the shared 64x128 bf16
//    weight tile into LDS (tensor_load_to_lds + s_wait_tensorcnt), all waves
//    consume it via ds_load_b128 B-fragments. 2 waves/WG, 48 KB LDS.
// ---------------------------------------------------------------------------
__global__ void ffn_kernel(const bf16_t* __restrict__ hb,
                           const bf16_t* __restrict__ wT,
                           const float* __restrict__ b1,
                           const float* __restrict__ b2, int blk,
                           const int* __restrict__ ctl, float* __restrict__ h) {
  if (!ctl[2]) return;
  int op = ctl[0];
  const bf16_t* W1T = wT + (size_t)4 * D * D;  // [H][D]
  const bf16_t* W2T = W1T + (size_t)D * H;     // [D][H]
  const float* B1 = b1 + ((size_t)blk * NOPS + op) * H;
  const float* B2 = b2 + ((size_t)blk * NOPS + op) * D;
  __shared__ bf16_t t_lds[2][16][H];  // 32 KB
  __shared__ bf16_t wbuf[64][D];      // 16 KB staged weight tile
  int lane = threadIdx.x & 31;
  int w = threadIdx.x >> 5;  // 2 waves / WG
  int gw = blockIdx.x * 2 + w;
  int row0 = gw * 16;
  int g = lane >> 4, n = lane & 15;

  // GEMM1: t = relu(hb @ W1 + b1); stage 64 W1 rows per n-group via TDM
  for (int ngrp = 0; ngrp < H / 64; ++ngrp) {
    int nbase = ngrp * 64;
    if (threadIdx.x == 0) {
      tdm_load_tile_2d(&wbuf[0][0], W1T + (size_t)nbase * D, D, H, D, 64, D);
      __builtin_amdgcn_s_wait_tensorcnt(0);
    }
    __syncthreads();
    v8f c[4] = {};
#pragma unroll
    for (int kc = 0; kc < D; kc += 32) {
      v16bf a = load_a_bf16(hb, D, row0, kc, lane);
#pragma unroll
      for (int t = 0; t < 4; ++t)
        c[t] = wmma_bf16(a, load_bT_bf16(&wbuf[0][0], D, kc, t * 16, lane),
                         c[t]);
    }
#pragma unroll
    for (int t = 0; t < 4; ++t) {
      float bias = B1[nbase + t * 16 + n];
#pragma unroll
      for (int r = 0; r < 8; ++r) {
        float z = c[t][r] + bias;
        t_lds[w][r + g * 8][nbase + t * 16 + n] = (bf16_t)(z > 0.f ? z : 0.f);
      }
    }
    __syncthreads();  // protect wbuf before next stage
  }
  // GEMM2: h = t @ W2 + b2; stage 64x128 W2T sub-tiles via TDM
  for (int ngrp = 0; ngrp < D / 64; ++ngrp) {
    int nbase = ngrp * 64;
    v8f c[4] = {};
    for (int kgrp = 0; kgrp < H / 128; ++kgrp) {
      if (threadIdx.x == 0) {
        tdm_load_tile_2d(&wbuf[0][0], W2T + (size_t)nbase * H + kgrp * 128, H,
                         D, 128, 64, H);
        __builtin_amdgcn_s_wait_tensorcnt(0);
      }
      __syncthreads();
#pragma unroll
      for (int kc = 0; kc < 128; kc += 32) {
        v16bf a = load_a_bf16(&t_lds[w][0][0], H, 0, kgrp * 128 + kc, lane);
#pragma unroll
        for (int t = 0; t < 4; ++t)
          c[t] = wmma_bf16(a, load_bT_bf16(&wbuf[0][0], 128, kc, t * 16, lane),
                           c[t]);
      }
      __syncthreads();
    }
#pragma unroll
    for (int t = 0; t < 4; ++t) {
      float bias = B2[nbase + t * 16 + n];
#pragma unroll
      for (int r = 0; r < 8; ++r)
        h[(size_t)(row0 + r + g * 8) * D + nbase + t * 16 + n] =
            c[t][r] + bias;
    }
  }
}

// ---------------------------------------------------------------------------
// host driver: 33 sequential routed blocks
// ---------------------------------------------------------------------------
extern "C" void kernel_launch(void* const* d_in, const int* in_sizes, int n_in,
                              void* d_out, int out_size, void* d_ws,
                              size_t ws_size, hipStream_t stream) {
  const float* x = (const float*)d_in[0];
  const float* ffn_w1 = (const float*)d_in[1];
  const float* ffn_b1 = (const float*)d_in[2];
  const float* ffn_w2 = (const float*)d_in[3];
  const float* ffn_b2 = (const float*)d_in[4];
  const float* wq = (const float*)d_in[5];
  const float* wk = (const float*)d_in[6];
  const float* wv = (const float*)d_in[7];
  const float* wo = (const float*)d_in[8];
  const unsigned char* am = (const unsigned char*)d_in[9];
  const unsigned char* fm = (const unsigned char*)d_in[10];

  char* ws = (char*)d_ws;
  int* ctl = (int*)ws;
  size_t ofs = 256;
  const size_t f32_bytes = (size_t)BS * D * sizeof(float);    // 4 MB
  const size_t bf16_bytes = (size_t)BS * D * sizeof(bf16_t);  // 2 MB
  float* h = (float*)(ws + ofs); ofs += f32_bytes;
  bf16_t* hb = (bf16_t*)(ws + ofs); ofs += bf16_bytes;
  bf16_t* qb = (bf16_t*)(ws + ofs); ofs += bf16_bytes;
  bf16_t* kb = (bf16_t*)(ws + ofs); ofs += bf16_bytes;
  bf16_t* vTb = (bf16_t*)(ws + ofs); ofs += bf16_bytes;
  bf16_t* ob = (bf16_t*)(ws + ofs); ofs += bf16_bytes;
  bf16_t* wT = (bf16_t*)(ws + ofs);
  ofs += (size_t)(4 * D * D + 2 * D * H) * sizeof(bf16_t);  // 384 KB

  (void)hipMemcpyAsync(h, x, f32_bytes, hipMemcpyDeviceToDevice, stream);

  for (int blk = 0; blk < NB; ++blk) {
    route_kernel<<<1, 32, 0, stream>>>(h, am, fm, blk, ctl);
    convert_h_kernel<<<BS * D / 4 / 256, 256, 0, stream>>>(h, hb);
    convert_w_kernel<<<(4 * D * D + 2 * D * H) / 256, 256, 0, stream>>>(
        wq, wk, wv, wo, ffn_w1, ffn_w2, blk, ctl, wT);
    qkv_kernel<<<384, 256, 0, stream>>>(hb, wT, ctl, qb, kb, vTb);
    attn_kernel<<<64, 256, 0, stream>>>(qb, kb, vTb, ctl, ob);
    outproj_kernel<<<128, 256, 0, stream>>>(ob, wT, ctl, h, hb);
    ffn_kernel<<<256, 64, 0, stream>>>(hb, wT, ffn_b1, ffn_b2, blk, ctl, h);
  }
  (void)hipMemcpyAsync(d_out, h, f32_bytes, hipMemcpyDeviceToDevice, stream);
}
